// RadarSparseBackboneDop_266287972908
// MI455X (gfx1250) — compile-verified, matrix operand order checked by reading, not run on hardware
//
#include <hip/hip_runtime.h>
#include <hip/hip_bf16.h>

// ---------------------------------------------------------------------------
// RadarSparseBackbone on MI455X (gfx1250): every conv is an implicit GEMM on
// v_wmma_f32_16x16x32_f16. One wave32 per 16x16 output tile; K tiled by 32
// (all channel counts are multiples of 32, so a K-tile never crosses a conv
// tap and fragment loads are two contiguous b128 per lane).
// Round-2 changes: zero-row pointer trick (branchless inner loop, no EXEC
// save/restore per K step) and compile-time K trip counts (templated CIN,
// split BEV K loop) so the scheduler can overlap load clauses with WMMA.
// ---------------------------------------------------------------------------

typedef _Float16 v16h __attribute__((ext_vector_type(16)));
typedef _Float16 v8h  __attribute__((ext_vector_type(8)));
typedef float    v8f  __attribute__((ext_vector_type(8)));

__device__ __forceinline__ v16h frag_load(const _Float16* __restrict__ p) {
  // CDNA5 16-bit fragment striping: this lane holds K = koff..koff+7 and
  // koff+16..koff+23 (caller already folded koff into p). Two 16B loads.
  v8h lo = *(const v8h*)(p);
  v8h hi = *(const v8h*)(p + 16);
  v16h r;
#pragma unroll
  for (int j = 0; j < 8; ++j) { r[j] = lo[j]; r[j + 8] = hi[j]; }
  return r;
}

__device__ __forceinline__ v8f wmma16(v16h a, v16h b, v8f c) {
  // 8 args: (neg_a, A, neg_b, B, c_mod, C, reuse_a, reuse_b)
  return __builtin_amdgcn_wmma_f32_16x16x32_f16(false, a, false, b, (short)0, c,
                                                false, false);
}

// ---------------------------------------------------------------------------
// Small prep kernels
// ---------------------------------------------------------------------------

// f32 (3,3,3,cin,cout) -> f16 [tap][cout][cin]  (transpose for contiguous B)
__global__ void k_wconv_enc(const float* __restrict__ w, _Float16* __restrict__ wT,
                            int cin, int cout) {
  int i = blockIdx.x * blockDim.x + threadIdx.x;
  int tot = 27 * cin * cout;
  if (i >= tot) return;
  int tap = i / (cout * cin);
  int r = i - tap * cout * cin;
  int n = r / cin;
  int ci = r - n * cin;
  wT[i] = (_Float16)w[((size_t)tap * cin + ci) * cout + n];
}

// BEV weight (k,k,256,2560) is already [tap][cout][cin]: plain f32->f16 copy
__global__ void k_f32_to_f16(const float* __restrict__ a, _Float16* __restrict__ b,
                             int n) {
  int i = blockIdx.x * blockDim.x + threadIdx.x;
  if (i < n) b[i] = (_Float16)a[i];
}

// Fold BN (+ optional conv bias) into per-channel scale/shift.
__global__ void k_bn_prep(const float* __restrict__ g, const float* __restrict__ b,
                          const float* __restrict__ m, const float* __restrict__ v,
                          const float* __restrict__ cbias,
                          float* __restrict__ scale, float* __restrict__ shift, int C) {
  int c = blockIdx.x * blockDim.x + threadIdx.x;
  if (c >= C) return;
  float s = g[c] * rsqrtf(v[c] + 1e-5f);
  float sh = b[c] - m[c] * s;
  if (cbias) sh += cbias[c] * s;
  scale[c] = s;
  shift[c] = sh;
}

// Scatter points: h[z,y,x,:] = feats(4) @ w_in(4,64); mark occupancy mask.
__global__ void k_scatter(const float* __restrict__ feats, const int* __restrict__ sidx,
                          const float* __restrict__ w_in, _Float16* __restrict__ h,
                          unsigned char* __restrict__ mask, int npts) {
  int i = blockIdx.x * blockDim.x + threadIdx.x;
  if (i >= npts * 64) return;
  int p = i >> 6;
  int c = i & 63;
  const int* id = sidx + (size_t)p * 4;
  int vox = (id[1] * 80 + id[2]) * 176 + id[3];
  const float* f = feats + (size_t)p * 4;
  float val = f[0] * w_in[c] + f[1] * w_in[64 + c] + f[2] * w_in[128 + c] +
              f[3] * w_in[192 + c];
  h[(size_t)vox * 64 + c] = (_Float16)val;
  if (c == 0) mask[vox] = 1;
}

// mask_out[vox] = OR over 3x3x3 window of mask_in (strided)
__global__ void k_mask_prop(const unsigned char* __restrict__ mi,
                            unsigned char* __restrict__ mo,
                            int Zi, int Yi, int Xi, int Zo, int Yo, int Xo, int stride) {
  int i = blockIdx.x * blockDim.x + threadIdx.x;
  int M = Zo * Yo * Xo;
  if (i >= M) return;
  int zo = i / (Yo * Xo);
  int r = i - zo * Yo * Xo;
  int yo = r / Xo, xo = r - yo * Xo;
  unsigned char v = 0;
  for (int dz = -1; dz <= 1; ++dz)
    for (int dy = -1; dy <= 1; ++dy)
      for (int dx = -1; dx <= 1; ++dx) {
        int zi = zo * stride + dz, yi = yo * stride + dy, xi = xo * stride + dx;
        if ((unsigned)zi < (unsigned)Zi && (unsigned)yi < (unsigned)Yi &&
            (unsigned)xi < (unsigned)Xi)
          v |= mi[((size_t)zi * Yi + yi) * Xi + xi];
      }
  mo[i] = v ? 1 : 0;
}

// [z][y][x][c] f16 -> [y][x][c*Zl+z] f16 (z-embed pack: BEV A rows contiguous)
__global__ void k_zembed(const _Float16* __restrict__ in, _Float16* __restrict__ out,
                         int Zl, int Yl, int Xl, int C) {
  int i = blockIdx.x * blockDim.x + threadIdx.x;
  int CZ = C * Zl;
  int tot = Yl * Xl * CZ;
  if (i >= tot) return;
  int pix = i / CZ;
  int clin = i - pix * CZ;
  int c = clin / Zl;
  int z = clin - c * Zl;
  int y = pix / Xl, x = pix - y * Xl;
  out[i] = in[(((size_t)z * Yl + y) * Xl + x) * C + c];
}

// ---------------------------------------------------------------------------
// 3x3x3 conv as implicit GEMM: D[M=out-voxels, N=Cout] over K=27*CIN.
// CIN/STRIDE are compile-time so the K loop fully unrolls (load clauses can
// run ahead of the WMMAs). Out-of-bounds taps read a zero-filled row instead
// of predicating loads -> branchless inner loop, EXEC stays all-ones.
// Fused epilogue: relu(acc*scale + shift) * mask -> f16.
// ---------------------------------------------------------------------------
template <int CIN, int STRIDE>
__global__ void __launch_bounds__(32)
k_conv3d_wmma(const _Float16* __restrict__ in, const _Float16* __restrict__ wT,
              const float* __restrict__ scale, const float* __restrict__ shift,
              const unsigned char* __restrict__ mask,
              const _Float16* __restrict__ zrow, _Float16* __restrict__ out,
              int Zi, int Yi, int Xi, int Zo, int Yo, int Xo, int Cout) {
  const int lane = threadIdx.x;
  const int mt = blockIdx.x, nt = blockIdx.y;
  const int M = Zo * Yo * Xo;
  const int row = mt * 16 + (lane & 15);
  const bool rv = row < M;
  int zo = 0, yo = 0, xo = 0;
  if (rv) {
    zo = row / (Yo * Xo);
    int r = row - zo * Yo * Xo;
    yo = r / Xo;
    xo = r - yo * Xo;
  }
  const int koff = (lane >> 4) * 8;          // lanes 16-31 hold K 8-15 / 24-31
  const int n = nt * 16 + (lane & 15);
  v8f acc;
#pragma unroll
  for (int j = 0; j < 8; ++j) acc[j] = 0.f;

  for (int tap = 0; tap < 27; ++tap) {
    const int kz = tap / 9, r9 = tap - kz * 9;
    const int ky = r9 / 3, kx = r9 - ky * 3;
    const int zi = zo * STRIDE + kz - 1;
    const int yi = yo * STRIDE + ky - 1;
    const int xi = xo * STRIDE + kx - 1;
    const bool av = rv && (unsigned)zi < (unsigned)Zi &&
                    (unsigned)yi < (unsigned)Yi && (unsigned)xi < (unsigned)Xi;
    const _Float16* arow =
        av ? in + (((size_t)(unsigned)zi * Yi + (unsigned)yi) * Xi + (unsigned)xi) * CIN + koff
           : zrow + koff;
    const _Float16* brow = wT + ((size_t)tap * Cout + n) * CIN + koff;
#pragma unroll
    for (int c0 = 0; c0 < CIN; c0 += 32)
      acc = wmma16(frag_load(arow + c0), frag_load(brow + c0), acc);
  }
  const float s = scale[n], sh = shift[n];
  const int mb = (lane >> 4) * 8;            // C layout: vgpr v -> M = mb+v
#pragma unroll
  for (int v = 0; v < 8; ++v) {
    const int r = mt * 16 + mb + v;
    if (r < M) {
      float val = acc[v] * s + sh;
      val = val > 0.f ? val : 0.f;
      val *= (float)mask[r];
      out[(size_t)r * Cout + n] = (_Float16)val;
    }
  }
}

// ---------------------------------------------------------------------------
// BEV i=0: k=3, s=1, p=1 "transposed" conv == 3x3 conv over the 80x176 plane.
// K loop split at the pw/dop boundary (no per-iteration source select);
// OOB taps read the zero row.
// ---------------------------------------------------------------------------
__global__ void __launch_bounds__(32)
k_bev_k3(const _Float16* __restrict__ zpw, const _Float16* __restrict__ zdop,
         const _Float16* __restrict__ wT, const float* __restrict__ scale,
         const float* __restrict__ shift, const _Float16* __restrict__ zrow,
         float* __restrict__ out, int chBase) {
  const int Xb = 176;
  const int lane = threadIdx.x;
  const int mt = blockIdx.x, nt = blockIdx.y;
  const int row = mt * 16 + (lane & 15);            // 14080 = 880*16, always valid
  const int oy = row / Xb, ox = row - oy * Xb;
  const int koff = (lane >> 4) * 8;
  const int n = nt * 16 + (lane & 15);
  v8f acc;
#pragma unroll
  for (int j = 0; j < 8; ++j) acc[j] = 0.f;

  for (int tap = 0; tap < 9; ++tap) {
    const int ky = tap / 3, kx = tap - ky * 3;
    const int iy = oy + 1 - ky, ix = ox + 1 - kx;   // flipped kernel (transpose)
    const bool av = (unsigned)iy < 80u && (unsigned)ix < 176u;
    const size_t pix = (size_t)(unsigned)iy * Xb + (unsigned)ix;
    const _Float16* ar = av ? zpw + pix * 1280 + koff : zrow + koff;
    const _Float16* ar2 = av ? zdop + pix * 1280 + koff : zrow + koff;
    const _Float16* brow = wT + ((size_t)tap * 256 + n) * 2560 + koff;
#pragma unroll 4
    for (int c0 = 0; c0 < 1280; c0 += 32)
      acc = wmma16(frag_load(ar + c0), frag_load(brow + c0), acc);
#pragma unroll 4
    for (int c0 = 0; c0 < 1280; c0 += 32)
      acc = wmma16(frag_load(ar2 + c0), frag_load(brow + 1280 + c0), acc);
  }
  const float s = scale[n], sh = shift[n];
  const int mb = (lane >> 4) * 8;
#pragma unroll
  for (int v = 0; v < 8; ++v) {
    const int r = mt * 16 + mb + v;
    const int ry = r / Xb, rx = r - ry * Xb;
    float val = acc[v] * s + sh;
    val = val > 0.f ? val : 0.f;
    out[(size_t)(chBase + n) * 14080 + (size_t)ry * Xb + rx] = val;
  }
}

// ---------------------------------------------------------------------------
// BEV i=1,2: k==s deconv. GEMM over input pixels per tap (grid.z), each
// output pixel receives exactly one tap; scatter in the epilogue.
// ---------------------------------------------------------------------------
__global__ void __launch_bounds__(32)
k_bev_up(const _Float16* __restrict__ zpw, const _Float16* __restrict__ zdop,
         const _Float16* __restrict__ wT, const float* __restrict__ scale,
         const float* __restrict__ shift, const _Float16* __restrict__ zrow,
         float* __restrict__ out, int chBase, int Yi, int Xi, int kk, int S) {
  const int lane = threadIdx.x;
  const int mt = blockIdx.x, nt = blockIdx.y, tap = blockIdx.z;
  const int ty = tap / kk, tx = tap - ty * kk;
  const int Min = Yi * Xi;
  const int row = mt * 16 + (lane & 15);
  const bool rv = row < Min;
  const int koff = (lane >> 4) * 8;
  const int n = nt * 16 + (lane & 15);
  const _Float16* ar = rv ? zpw + (size_t)row * 1280 + koff : zrow + koff;
  const _Float16* ar2 = rv ? zdop + (size_t)row * 1280 + koff : zrow + koff;
  const _Float16* brow = wT + ((size_t)tap * 256 + n) * 2560 + koff;
  v8f acc;
#pragma unroll
  for (int j = 0; j < 8; ++j) acc[j] = 0.f;

#pragma unroll 4
  for (int c0 = 0; c0 < 1280; c0 += 32)
    acc = wmma16(frag_load(ar + c0), frag_load(brow + c0), acc);
#pragma unroll 4
  for (int c0 = 0; c0 < 1280; c0 += 32)
    acc = wmma16(frag_load(ar2 + c0), frag_load(brow + 1280 + c0), acc);

  const float s = scale[n], sh = shift[n];
  const int mb = (lane >> 4) * 8;
#pragma unroll
  for (int v = 0; v < 8; ++v) {
    const int r = mt * 16 + mb + v;
    if (r < Min) {
      const int ry = r / Xi, rx = r - ry * Xi;
      const int oy = ry * S + ty, ox = rx * S + tx;
      float val = acc[v] * s + sh;
      val = val > 0.f ? val : 0.f;
      out[(size_t)(chBase + n) * 14080 + (size_t)oy * 176 + ox] = val;
    }
  }
}

// ---------------------------------------------------------------------------
// Host orchestration
// ---------------------------------------------------------------------------
extern "C" void kernel_launch(void* const* d_in, const int* in_sizes, int n_in,
                              void* d_out, int out_size, void* d_ws, size_t ws_size,
                              hipStream_t stream) {
  (void)n_in; (void)out_size; (void)ws_size;
  static const int Z0 = 20, Y0 = 80, X0 = 176;
  static const int Z1 = 10, Y1 = 40, X1 = 88;
  static const int Z2 = 5, Y2 = 20, X2 = 44;
  const size_t V0 = (size_t)Z0 * Y0 * X0;      // 281600
  const size_t V1 = (size_t)Z1 * Y1 * X1;      // 35200
  const size_t V2 = (size_t)Z2 * Y2 * X2;      // 4400

  const float* feats[2] = {(const float*)d_in[0], (const float*)d_in[1]};
  const int* sidx = (const int*)d_in[2];
  const int NP = in_sizes[2] / 4;

  // pytree flatten order (dict keys sorted): params = {layers:[...], w_in}
  // layer leaves: bn_a{b,g,m,v}, bn_b{b,g,m,v}, bn_sp{b,g,m,v}, w_a, w_b, w_sp
  // params_bev[i]: bias, bn{b,g,m,v}, w
  const int BASE[2] = {3, 49};
  const int B_BEV = 95;

  // ---- workspace bump allocator ----
  char* ws = (char*)d_ws;
  size_t off = 0;
  auto alloc = [&](size_t b) -> char* {
    char* p = ws + off;
    off += (b + 255) & ~(size_t)255;
    return p;
  };

  _Float16* bufA = (_Float16*)alloc(V0 * 64 * 2);
  _Float16* bufB = (_Float16*)alloc(V0 * 64 * 2);
  _Float16* out0[2], *out1[2], *out2[2];
  for (int e = 0; e < 2; ++e) {
    out0[e] = (_Float16*)alloc(V0 * 64 * 2);
    out1[e] = (_Float16*)alloc(V1 * 128 * 2);
    out2[e] = (_Float16*)alloc(V2 * 256 * 2);
  }
  unsigned char* mIn = (unsigned char*)alloc(V0);
  unsigned char* m0 = (unsigned char*)alloc(V0);
  unsigned char* m1 = (unsigned char*)alloc(V1);
  unsigned char* m2 = (unsigned char*)alloc(V2);
  _Float16* zrow = (_Float16*)alloc(2600 * 2);   // shared zero row for OOB taps

  const int CIN[9] = {64, 64, 64, 64, 128, 128, 128, 256, 256};
  const int COUT[9] = {64, 64, 64, 128, 128, 128, 256, 256, 256};
  const int WITEM[3] = {14, 12, 13};  // sp, a, b -> w leaf offset
  const int BNOFF[3] = {8, 0, 4};     // sp, a, b -> bn leaf base {b,g,m,v}

  _Float16* encW[2][9];
  float *encS[2][9], *encH[2][9];
  for (int e = 0; e < 2; ++e)
    for (int j = 0; j < 9; ++j) {
      encW[e][j] = (_Float16*)alloc((size_t)27 * CIN[j] * COUT[j] * 2);
      encS[e][j] = (float*)alloc(COUT[j] * 4);
      encH[e][j] = (float*)alloc(COUT[j] * 4);
    }

  _Float16* zemb1[2], *zemb2[2];
  for (int e = 0; e < 2; ++e) {
    zemb1[e] = (_Float16*)alloc((size_t)Y1 * X1 * 1280 * 2);
    zemb2[e] = (_Float16*)alloc((size_t)Y2 * X2 * 1280 * 2);
  }
  const int BK[3] = {3, 2, 4};
  _Float16* bevW[3];
  float *bevS[3], *bevH[3];
  for (int i = 0; i < 3; ++i) {
    bevW[i] = (_Float16*)alloc((size_t)BK[i] * BK[i] * 256 * 2560 * 2);
    bevS[i] = (float*)alloc(256 * 4);
    bevH[i] = (float*)alloc(256 * 4);
  }

  // ---- weight / BN prep ----
  for (int e = 0; e < 2; ++e) {
    for (int j = 0; j < 9; ++j) {
      int layer = j / 3, which = j % 3;
      int lb = BASE[e] + layer * 15;
      const float* w = (const float*)d_in[lb + WITEM[which]];
      int tot = 27 * CIN[j] * COUT[j];
      k_wconv_enc<<<(tot + 255) / 256, 256, 0, stream>>>(w, encW[e][j], CIN[j], COUT[j]);
      const float* bb = (const float*)d_in[lb + BNOFF[which] + 0];
      const float* bg = (const float*)d_in[lb + BNOFF[which] + 1];
      const float* bm = (const float*)d_in[lb + BNOFF[which] + 2];
      const float* bv = (const float*)d_in[lb + BNOFF[which] + 3];
      k_bn_prep<<<(COUT[j] + 63) / 64, 64, 0, stream>>>(bg, bb, bm, bv, nullptr,
                                                        encS[e][j], encH[e][j], COUT[j]);
    }
  }
  for (int i = 0; i < 3; ++i) {
    int lb = B_BEV + i * 6;
    const float* bias = (const float*)d_in[lb + 0];
    const float* bb = (const float*)d_in[lb + 1];
    const float* bg = (const float*)d_in[lb + 2];
    const float* bm = (const float*)d_in[lb + 3];
    const float* bv = (const float*)d_in[lb + 4];
    const float* w = (const float*)d_in[lb + 5];
    int tot = BK[i] * BK[i] * 256 * 2560;
    k_f32_to_f16<<<(tot + 255) / 256, 256, 0, stream>>>(w, bevW[i], tot);
    k_bn_prep<<<4, 64, 0, stream>>>(bg, bb, bm, bv, bias, bevS[i], bevH[i], 256);
  }

  // ---- masks (shared between encoders: same indices) + zero row ----
  hipMemsetAsync(zrow, 0, 2600 * 2, stream);
  hipMemsetAsync(mIn, 0, V0, stream);
  hipMemsetAsync(bufA, 0, V0 * 64 * 2, stream);
  k_scatter<<<(NP * 64 + 255) / 256, 256, 0, stream>>>(
      feats[0], sidx, (const float*)d_in[BASE[0] + 45], bufA, mIn, NP);
  k_mask_prop<<<((int)V0 + 255) / 256, 256, 0, stream>>>(mIn, m0, Z0, Y0, X0, Z0, Y0, X0, 1);
  k_mask_prop<<<((int)V1 + 255) / 256, 256, 0, stream>>>(m0, m1, Z0, Y0, X0, Z1, Y1, X1, 2);
  k_mask_prop<<<((int)V2 + 255) / 256, 256, 0, stream>>>(m1, m2, Z1, Y1, X1, Z2, Y2, X2, 2);

  // ---- encoders ----
  struct Cfg { int Zi, Yi, Xi, Zo, Yo, Xo; };
  const Cfg CFG[9] = {
      {Z0, Y0, X0, Z0, Y0, X0}, {Z0, Y0, X0, Z0, Y0, X0}, {Z0, Y0, X0, Z0, Y0, X0},
      {Z0, Y0, X0, Z1, Y1, X1}, {Z1, Y1, X1, Z1, Y1, X1}, {Z1, Y1, X1, Z1, Y1, X1},
      {Z1, Y1, X1, Z2, Y2, X2}, {Z2, Y2, X2, Z2, Y2, X2}, {Z2, Y2, X2, Z2, Y2, X2}};
  unsigned char* MASK[9] = {m0, m0, m0, m1, m1, m1, m2, m2, m2};

  for (int e = 0; e < 2; ++e) {
    if (e == 1) {  // pw already scattered into bufA above
      hipMemsetAsync(bufA, 0, V0 * 64 * 2, stream);
      k_scatter<<<(NP * 64 + 255) / 256, 256, 0, stream>>>(
          feats[1], sidx, (const float*)d_in[BASE[1] + 45], bufA, mIn, NP);
    }
    const _Float16* IN[9] = {bufA, bufB, bufA, out0[e], bufA, bufB, out1[e], bufA, bufB};
    _Float16* OUT[9] = {bufB, bufA, out0[e], bufA, bufB, out1[e], bufA, bufB, out2[e]};
    for (int j = 0; j < 9; ++j) {
      const Cfg& c = CFG[j];
      int M = c.Zo * c.Yo * c.Xo;
      dim3 grid((unsigned)((M + 15) / 16), (unsigned)(COUT[j] / 16));
      switch (j) {
        case 0: case 1: case 2:
          k_conv3d_wmma<64, 1><<<grid, 32, 0, stream>>>(
              IN[j], encW[e][j], encS[e][j], encH[e][j], MASK[j], zrow, OUT[j],
              c.Zi, c.Yi, c.Xi, c.Zo, c.Yo, c.Xo, COUT[j]);
          break;
        case 3:
          k_conv3d_wmma<64, 2><<<grid, 32, 0, stream>>>(
              IN[j], encW[e][j], encS[e][j], encH[e][j], MASK[j], zrow, OUT[j],
              c.Zi, c.Yi, c.Xi, c.Zo, c.Yo, c.Xo, COUT[j]);
          break;
        case 4: case 5:
          k_conv3d_wmma<128, 1><<<grid, 32, 0, stream>>>(
              IN[j], encW[e][j], encS[e][j], encH[e][j], MASK[j], zrow, OUT[j],
              c.Zi, c.Yi, c.Xi, c.Zo, c.Yo, c.Xo, COUT[j]);
          break;
        case 6:
          k_conv3d_wmma<128, 2><<<grid, 32, 0, stream>>>(
              IN[j], encW[e][j], encS[e][j], encH[e][j], MASK[j], zrow, OUT[j],
              c.Zi, c.Yi, c.Xi, c.Zo, c.Yo, c.Xo, COUT[j]);
          break;
        default:
          k_conv3d_wmma<256, 1><<<grid, 32, 0, stream>>>(
              IN[j], encW[e][j], encS[e][j], encH[e][j], MASK[j], zrow, OUT[j],
              c.Zi, c.Yi, c.Xi, c.Zo, c.Yo, c.Xo, COUT[j]);
          break;
      }
    }
  }

  // ---- z-embed packs (bufA/bufB are free now; reuse for level-0 planes) ----
  _Float16* zemb0[2] = {bufA, bufB};
  for (int e = 0; e < 2; ++e) {
    int t0 = (int)(V0 * 64), t1 = (int)(V1 * 128), t2 = (int)(V2 * 256);
    k_zembed<<<(t0 + 255) / 256, 256, 0, stream>>>(out0[e], zemb0[e], Z0, Y0, X0, 64);
    k_zembed<<<(t1 + 255) / 256, 256, 0, stream>>>(out1[e], zemb1[e], Z1, Y1, X1, 128);
    k_zembed<<<(t2 + 255) / 256, 256, 0, stream>>>(out2[e], zemb2[e], Z2, Y2, X2, 256);
  }

  // ---- BEV heads -> d_out (1,768,80,176) f32 ----
  float* outp = (float*)d_out;
  {
    dim3 g0(880, 16);
    k_bev_k3<<<g0, 32, 0, stream>>>(zemb0[0], zemb0[1], bevW[0], bevS[0], bevH[0],
                                    zrow, outp, 0);
    dim3 g1(220, 16, 4);
    k_bev_up<<<g1, 32, 0, stream>>>(zemb1[0], zemb1[1], bevW[1], bevS[1], bevH[1],
                                    zrow, outp, 256, Y1, X1, 2, 2);
    dim3 g2(55, 16, 16);
    k_bev_up<<<g2, 32, 0, stream>>>(zemb2[0], zemb2[1], bevW[2], bevS[2], bevH[2],
                                    zrow, outp, 512, Y2, X2, 4, 4);
  }
}
